// layer_50087908606304
// MI455X (gfx1250) — matrix operand order repaired and checked by reading, not compiled, not run on hardware
//
#include <hip/hip_runtime.h>
#include <math.h>
#include <stdint.h>

// ---------------------------------------------------------------------------
// Autoregressive LSTM decode for MI455X (gfx1250, wave32, WMMA).
// B=32, S=48, V=32000, E=H=512.
// GEMMs use v_wmma_f32_16x16x32_bf16: fp32 weights are loaded from L2 and
// converted to bf16 in registers (weights fit in 192MB L2; HBM traffic is
// dominated by the 197MB d_out stream). fp32 accumulation throughout.
// ---------------------------------------------------------------------------

typedef __bf16 bf16_t;
typedef __attribute__((ext_vector_type(16))) __bf16 v16bf;
typedef __attribute__((ext_vector_type(8)))  float  v8f;

#define BB 32      // batch
#define SS 48      // seq length
#define VV 32000   // vocab
#define EE 512     // embed dim
#define HH 512     // hidden dim

// ---- WMMA fragment loaders (fp32 memory -> bf16 fragments) ----------------
// A-matrix 16x32 bf16 (ISA 7.12.2): lane 0-15 -> M=lane, K base 0; lane 16-31
// -> M=lane-16, K base +8 for VGPRs 0-3; VGPRs 4-7 hold K=16..31 same split.
__device__ inline v16bf load_a_frag(const float* __restrict__ P, int ld,
                                    int mbase, int kb) {
    const int lane = threadIdx.x & 31;
    const int m    = mbase + (lane & 15);
    const int khi  = (lane >> 4) * 8;          // lanes 16-31: +8 within halves
    const float* row = P + (size_t)m * ld + kb;
    v16bf a = {};
#pragma unroll
    for (int p = 0; p < 8; ++p) {
        const int k = ((p < 4) ? (2 * p) : (16 + 2 * (p - 4))) + khi;
        const float2 f = *(const float2*)(row + k);
        a[2 * p]     = (bf16_t)f.x;
        a[2 * p + 1] = (bf16_t)f.y;
    }
    return a;
}

// B-matrix 32x16 bf16: lane&15 -> N column; same K split as A: lanes 16-31
// +8 within each K half (VGPRs 0-3 K=0..15, VGPRs 4-7 K=16..31).
// W is row-major [out_features x ld]; B(k,n) = W[nbase+n, kb+k].
__device__ inline v16bf load_b_frag(const float* __restrict__ W, int ld,
                                    int nbase, int kb) {
    const int lane = threadIdx.x & 31;
    const int n    = nbase + (lane & 15);
    const int khi  = (lane >> 4) * 8;
    const float* row = W + (size_t)n * ld + kb + khi;
    v16bf b = {};
#pragma unroll
    for (int p = 0; p < 8; ++p) {
        const float2 f = *(const float2*)(row + ((p < 4) ? (2 * p) : (16 + 2 * (p - 4))));
        b[2 * p]     = (bf16_t)f.x;
        b[2 * p + 1] = (bf16_t)f.y;
    }
    return b;
}

__device__ inline v8f wmma_bf16(v16bf a, v16bf b, v8f c) {
    // (neg_a, A, neg_b, B, c_mod, C, reuse_a, reuse_b)
    return __builtin_amdgcn_wmma_f32_16x16x32_bf16(false, a, false, b,
                                                   (short)0, c, false, false);
}

__device__ inline float sigmoidf(float x) { return 1.0f / (1.0f + __expf(-x)); }

// ---- Kernel 1: gates = x@Wi.T + h@Wh.T + (bi+bh); fused LSTM cell ---------
// One wave per 16-column slice of H. Computes all 4 gates (i,f,g,o at column
// offsets g*H + j) for both 16-row M-tiles, then updates c and writes hnext.
// Reads hprev (all columns), writes hnext (own slice) -> ping-pong buffers.
__global__ __launch_bounds__(32)
void lstm_gates_cell(const float* __restrict__ x, const float* __restrict__ hprev,
                     float* __restrict__ c, float* __restrict__ hnext,
                     const float* __restrict__ Wi, const float* __restrict__ Wh,
                     const float* __restrict__ bi, const float* __restrict__ bh) {
    const int j0 = blockIdx.x * 16;
    v8f acc[4][2];
#pragma unroll
    for (int g = 0; g < 4; ++g) { acc[g][0] = {}; acc[g][1] = {}; }

#pragma unroll 1
    for (int ph = 0; ph < 2; ++ph) {
        const float* X = ph ? hprev : x;
        const float* W = ph ? Wh : Wi;
#pragma unroll 1
        for (int kb = 0; kb < EE; kb += 32) {
            const v16bf a0 = load_a_frag(X, EE, 0, kb);
            const v16bf a1 = load_a_frag(X, EE, 16, kb);
#pragma unroll
            for (int g = 0; g < 4; ++g) {
                const v16bf bf = load_b_frag(W, EE, g * HH + j0, kb);
                acc[g][0] = wmma_bf16(a0, bf, acc[g][0]);
                acc[g][1] = wmma_bf16(a1, bf, acc[g][1]);
            }
        }
    }

    const int lane = threadIdx.x & 31;
    const int n    = lane & 15;
    const int j    = j0 + n;
    float bsum[4];
#pragma unroll
    for (int g = 0; g < 4; ++g) bsum[g] = bi[g * HH + j] + bh[g * HH + j];

    const int mo = (lane >> 4) * 8;   // C/D layout: lanes 16-31 -> rows +8
#pragma unroll
    for (int t = 0; t < 2; ++t) {
#pragma unroll
        for (int v = 0; v < 8; ++v) {
            const int m   = t * 16 + mo + v;
            const float ig = sigmoidf(acc[0][t][v] + bsum[0]);
            const float fg = sigmoidf(acc[1][t][v] + bsum[1]);
            const float gg = tanhf  (acc[2][t][v] + bsum[2]);
            const float og = sigmoidf(acc[3][t][v] + bsum[3]);
            const float cc = fg * c[(size_t)m * HH + j] + ig * gg;
            c[(size_t)m * HH + j]     = cc;
            hnext[(size_t)m * HH + j] = og * tanhf(cc);
        }
    }
}

// ---- Kernel 2: logits = h @ Wo.T + bo  (32 x 32000), one wave / 16 cols ---
__global__ __launch_bounds__(32)
void vocab_logits(const float* __restrict__ h, const float* __restrict__ Wo,
                  const float* __restrict__ bo, float* __restrict__ logits) {
    const int n0 = blockIdx.x * 16;
    v8f acc0 = {}, acc1 = {};
#pragma unroll 1
    for (int kb = 0; kb < HH; kb += 32) {
        const v16bf a0 = load_a_frag(h, HH, 0, kb);
        const v16bf a1 = load_a_frag(h, HH, 16, kb);
        const v16bf bf = load_b_frag(Wo, HH, n0, kb);
        acc0 = wmma_bf16(a0, bf, acc0);
        acc1 = wmma_bf16(a1, bf, acc1);
    }
    const int lane = threadIdx.x & 31;
    const int n    = lane & 15;
    const int col  = n0 + n;
    const float bv = bo[col];
    const int mo   = (lane >> 4) * 8;
#pragma unroll
    for (int v = 0; v < 8; ++v) {
        logits[(size_t)(mo + v) * VV + col]      = acc0[v] + bv;
        logits[(size_t)(16 + mo + v) * VV + col] = acc1[v] + bv;
    }
}

// ---- Kernel 3: per-row log-sum-exp over vocab (online max/sum) ------------
__global__ __launch_bounds__(256)
void row_lse(const float* __restrict__ logits, float* __restrict__ lse) {
    const int b   = blockIdx.x;
    const int tid = threadIdx.x;
    const float* row = logits + (size_t)b * VV;
    float m = -INFINITY, s = 0.0f;
    for (int v = tid; v < VV; v += 256) {           // 32000/256 = 125, exact
        const float x  = row[v];
        const float nm = fmaxf(m, x);
        s = s * __expf(m - nm) + __expf(x - nm);
        m = nm;
    }
    __shared__ float sm[256], ssum[256];
    sm[tid] = m; ssum[tid] = s;
    __syncthreads();
    for (int w = 128; w > 0; w >>= 1) {
        if (tid < w) {
            const float m2 = sm[tid + w], s2 = ssum[tid + w];
            const float nm = fmaxf(sm[tid], m2);
            ssum[tid] = ssum[tid] * __expf(sm[tid] - nm) + s2 * __expf(m2 - nm);
            sm[tid] = nm;
        }
        __syncthreads();
    }
    if (tid == 0) lse[b] = sm[0] + __logf(ssum[0]);
}

// ---- Kernel 4: store masked log-probs, Gumbel-max sample, embed next x ----
__device__ inline uint32_t mix32(uint32_t h) {
    h ^= h >> 16; h *= 0x7feb352dU;
    h ^= h >> 15; h *= 0x846ca68bU;
    h ^= h >> 16; return h;
}

__global__ __launch_bounds__(256)
void sample_and_embed(const float* __restrict__ logits, const float* __restrict__ lse,
                      int* __restrict__ active, const float* __restrict__ emb,
                      float* __restrict__ x, float* __restrict__ out, int step) {
    const int b   = blockIdx.x;
    const int tid = threadIdx.x;
    const float* row = logits + (size_t)b * VV;
    const float L   = lse[b];
    const int   act = active[b];                  // active BEFORE this sample
    float* orow = out + ((size_t)b * SS + (size_t)step) * VV;

    float best = -INFINITY;
    int   bidx = 0;
    const uint32_t seed = mix32(((uint32_t)step << 8) ^ (uint32_t)b ^ 0xA5A5A5A5U);
    for (int v = tid; v < VV; v += 256) {
        const float lg = row[v];
        orow[v] = act ? (lg - L) : 0.0f;          // stored log-prob (masked)
        const uint32_t hsh = mix32(seed ^ (uint32_t)v);
        const float u = (float)(hsh >> 8) * (1.0f / 16777216.0f) + 1e-10f;
        const float g = -__logf(-__logf(u));      // Gumbel(0,1)
        const float sc = lg + g;                  // argmax == categorical(logp)
        if (sc > best) { best = sc; bidx = v; }
    }
    __shared__ float sb[256];
    __shared__ int   si[256];
    __shared__ int   stok;
    sb[tid] = best; si[tid] = bidx;
    __syncthreads();
    for (int w = 128; w > 0; w >>= 1) {
        if (tid < w && sb[tid + w] > sb[tid]) {
            sb[tid] = sb[tid + w]; si[tid] = si[tid + w];
        }
        __syncthreads();
    }
    if (tid == 0) {
        const int tok = si[0];
        stok = tok;
        active[b] = act && (tok != VV - 2);       // EOS check
    }
    __syncthreads();
    const int tok = stok;
    for (int i = tid; i < EE; i += 256)           // next-step input
        x[(size_t)b * EE + i] = emb[(size_t)tok * EE + i];
}

// ---- Init: x = encoded, h0 = 0, c = 0, active = 1 -------------------------
__global__ __launch_bounds__(256)
void init_state(const float* __restrict__ enc, float* __restrict__ x,
                float* __restrict__ h0, float* __restrict__ c,
                int* __restrict__ active) {
    const int i = blockIdx.x * 256 + threadIdx.x;
    if (i < BB * HH) { x[i] = enc[i]; h0[i] = 0.0f; c[i] = 0.0f; }
    if (i < BB) active[i] = 1;
}

// ---------------------------------------------------------------------------
extern "C" void kernel_launch(void* const* d_in, const int* in_sizes, int n_in,
                              void* d_out, int out_size, void* d_ws, size_t ws_size,
                              hipStream_t stream) {
    const float* enc = (const float*)d_in[0];   // [B,E]
    const float* emb = (const float*)d_in[1];   // [V,E]
    const float* Wi  = (const float*)d_in[2];   // [4H,E]
    const float* Wh  = (const float*)d_in[3];   // [4H,H]
    const float* bi  = (const float*)d_in[4];   // [4H]
    const float* bh  = (const float*)d_in[5];   // [4H]
    const float* Wo  = (const float*)d_in[6];   // [V,H]
    const float* bo  = (const float*)d_in[7];   // [V]
    float* out = (float*)d_out;                 // [B,S,V]

    // Carve workspace (all buffers we read are fully initialized below).
    char* ws = (char*)d_ws;
    size_t off = 0;
    auto carve = [&](size_t bytes) -> void* {
        void* p = ws + off;
        off = (off + bytes + 255) & ~(size_t)255;
        return p;
    };
    float* h0     = (float*)carve(sizeof(float) * BB * HH);
    float* h1     = (float*)carve(sizeof(float) * BB * HH);
    float* cbuf   = (float*)carve(sizeof(float) * BB * HH);
    float* xbuf   = (float*)carve(sizeof(float) * BB * EE);
    float* logits = (float*)carve(sizeof(float) * (size_t)BB * VV);
    float* lse    = (float*)carve(sizeof(float) * BB);
    int*   active = (int*)carve(sizeof(int) * BB);

    init_state<<<(BB * HH + 255) / 256, 256, 0, stream>>>(enc, xbuf, h0, cbuf, active);

    float* hp = h0;   // h_{t-1} (read by all gate blocks)
    float* hn = h1;   // h_t (written, slice-disjoint)
    for (int s = 0; s < SS; ++s) {
        lstm_gates_cell<<<HH / 16, 32, 0, stream>>>(xbuf, hp, cbuf, hn, Wi, Wh, bi, bh);
        vocab_logits<<<VV / 16, 32, 0, stream>>>(hn, Wo, bo, logits);
        row_lse<<<BB, 256, 0, stream>>>(logits, lse);
        sample_and_embed<<<BB, 256, 0, stream>>>(logits, lse, active, emb, xbuf, out, s);
        float* t = hp; hp = hn; hn = t;          // ping-pong h
    }
}
